// DeformUpSample_20564303413698
// MI455X (gfx1250) — matrix operand order repaired and checked by reading, not compiled, not run on hardware
//
#include <hip/hip_runtime.h>
#include <cstdint>

// Problem constants (match reference)
#define B_    2
#define C_    64
#define H_    128
#define W_    128
#define HW_   (H_*W_)
#define KK_   9
#define KTOT  576          // 64 channels * 9 taps
#define KSTEPS 18          // 576 / 32 (wmma f16 k=32)
#define M1TILES 7          // conv1: 108 rows -> 7 tiles of 16 (pad to 112)
#define OFFC  112          // off buffer padded channel count (108 used)
#define COLPITCH 584       // f16 per LDS row (576 + 8 pad to spread banks)
#define NCNT  131072.0f    // BN count per channel: B * 4 * H * W

typedef __attribute__((ext_vector_type(16))) _Float16 v16h;
typedef __attribute__((ext_vector_type(8)))  _Float16 v8h;
typedef __attribute__((ext_vector_type(8)))  float    v8f;

// ---------------------------------------------------------------------------
// Fragment loaders.
// A (16x32 f16, same K order as B): lane l -> row M = l%16; element e -> K =
//   (l/16)*16 + e. Weights are PRE-PACKED into this order, so
//   the load is 32 contiguous bytes per lane (two b128 global loads).
// B (32x16 f16): lane l -> col N = l%16; element e -> K = (l/16)*16 + e
//   (this file's own order, not the ISA table: lanes 0-15 hold K=0..15,
//   lanes 16-31 hold K=16..31). 16 contiguous f16 in our [n][k] LDS tile
//   (two ds_load_b128).
// ---------------------------------------------------------------------------
__device__ __forceinline__ v16h afrag(const _Float16* __restrict__ tileBase,
                                      int ks, int lane) {
  const _Float16* p = tileBase + (size_t)(ks * 32 + lane) * 16;
  v8h lo = *(const v8h*)p;
  v8h hi = *(const v8h*)(p + 8);
  return __builtin_shufflevector(lo, hi, 0,1,2,3,4,5,6,7,8,9,10,11,12,13,14,15);
}

__device__ __forceinline__ v16h bfrag(const _Float16* colp, int nt, int ks, int lane) {
  const _Float16* p = colp + (size_t)(nt * 16 + (lane & 15)) * COLPITCH
                           + ks * 32 + (lane >> 4) * 16;
  v8h lo = *(const v8h*)p;
  v8h hi = *(const v8h*)(p + 8);
  return __builtin_shufflevector(lo, hi, 0,1,2,3,4,5,6,7,8,9,10,11,12,13,14,15);
}

// ---------------------------------------------------------------------------
// Stage 0a: x (B,C,H,W) -> xt (B,H,W,C) so a bilinear tap is 64 contiguous f32
// ---------------------------------------------------------------------------
__global__ void k_transpose(const float* __restrict__ x, float* __restrict__ xt) {
  int i = blockIdx.x * 256 + threadIdx.x;            // ((b*H+h)*W+w)*64 + c
  if (i >= B_ * HW_ * C_) return;
  int c = i & 63;
  int w = (i >> 6) & 127;
  int h = (i >> 13) & 127;
  int b = i >> 20;
  xt[i] = x[(((size_t)(b * C_ + c) * H_ + h) * W_ + w)];
}

// ---------------------------------------------------------------------------
// Stage 0b: fold w_offset over the 4 tiled copies of x (Cin 256 -> 64) and
// pack into A-fragment order.  w1p[((mt*18+ks)*32+lane)*16+e]
// ---------------------------------------------------------------------------
__global__ void k_pack_w1(const float* __restrict__ w_offset, _Float16* __restrict__ w1p) {
  int i = blockIdx.x * 256 + threadIdx.x;
  if (i >= M1TILES * KSTEPS * 512) return;
  int e = i & 15, lane = (i >> 4) & 31;
  int ks = (i >> 9) % KSTEPS, mt = (i >> 9) / KSTEPS;
  int m = mt * 16 + (lane & 15);
  int k = ks * 32 + (lane >> 4) * 16 + e;
  int c = k / 9, p = k % 9;
  float v = 0.f;
  if (m < 108) {
    #pragma unroll
    for (int s = 0; s < 4; ++s)                       // fold over tiling
      v += w_offset[(size_t)(m * 256 + s * 64 + c) * 9 + p];
  }
  w1p[i] = (_Float16)v;
}

// ---------------------------------------------------------------------------
// Stage 0c: pack w_dcn (256,64,3,3) into per-(group,mtile) A fragments.
// ---------------------------------------------------------------------------
__global__ void k_pack_w2(const float* __restrict__ w_dcn, _Float16* __restrict__ w2p) {
  int i = blockIdx.x * 256 + threadIdx.x;
  if (i >= 4 * 4 * KSTEPS * 512) return;
  int e = i & 15, lane = (i >> 4) & 31;
  int ks = (i >> 9) % KSTEPS, rest = (i >> 9) / KSTEPS;
  int mt = rest & 3, g = rest >> 2;
  int o = mt * 16 + (lane & 15);
  int k = ks * 32 + (lane >> 4) * 16 + e;
  int c = k / 9, p = k % 9;
  w2p[i] = (_Float16)w_dcn[(size_t)(g * 64 + o) * KTOT + c * 9 + p];
}

// ---------------------------------------------------------------------------
// Stage 1: offset conv as implicit GEMM. One workgroup = one batch, 32 pixels.
// LDS col[32][576] built from shifted x (im2col), then 14 wmma tiles.
// off is padded to 112 channels so the writeout has no divergent guard.
// ---------------------------------------------------------------------------
__global__ __launch_bounds__(256) void k_offconv(const float* __restrict__ xt,
                                                 const _Float16* __restrict__ w1p,
                                                 const float* __restrict__ bias,
                                                 float* __restrict__ off) {
  __shared__ _Float16 col[32][COLPITCH];
  int wgid = blockIdx.x;                 // B * 512
  int b = wgid >> 9;
  int base = (wgid & 511) * 32;          // pixel base within HW (row-major)
  int h = base >> 7, w0 = base & 127;
  int tid = threadIdx.x, lane = tid & 31, wv = tid >> 5;

  // gather: 32 pixels * 9 taps = 288 (pix,p) pairs, 36 per wave;
  // lanes cover 64 channels (2 each) with one float2 load.
  for (int q = wv * 36; q < wv * 36 + 36; ++q) {
    int pix = q / 9, p = q % 9;
    int yy = h + p / 3 - 1;
    int xx = w0 + pix + p % 3 - 1;
    float2 v = make_float2(0.f, 0.f);
    if (yy >= 0 && yy < H_ && xx >= 0 && xx < W_)
      v = *(const float2*)(xt + (((size_t)b * HW_ + yy * W_ + xx) << 6) + 2 * lane);
    int c = 2 * lane;
    col[pix][c * 9 + p]       = (_Float16)v.x;
    col[pix][(c + 1) * 9 + p] = (_Float16)v.y;
  }
  __syncthreads();

  // GEMM: 7 m-tiles x 2 n-tiles = 14 tiles over 8 waves.
  // Two independent accumulator chains (even/odd k-steps) for WMMA ILP.
  for (int t = wv; t < 14; t += 8) {
    int mt = t % 7, nt = t / 7;
    v8f acc0 = {}, acc1 = {};
    const _Float16* aw = w1p + (size_t)mt * KSTEPS * 512;
    for (int ks = 0; ks < KSTEPS; ks += 2) {
      v16h a0 = afrag(aw, ks, lane);
      v16h b0 = bfrag(&col[0][0], nt, ks, lane);
      v16h a1 = afrag(aw, ks + 1, lane);
      v16h b1 = bfrag(&col[0][0], nt, ks + 1, lane);
      acc0 = __builtin_amdgcn_wmma_f32_16x16x32_f16(false, a0, false, b0,
                                                    (short)0, acc0, false, false);
      acc1 = __builtin_amdgcn_wmma_f32_16x16x32_f16(false, a1, false, b1,
                                                    (short)0, acc1, false, false);
    }
    int n = nt * 16 + (lane & 15);
    int half = lane >> 4;
    int hw = base + n;
    float* op = off + (((size_t)(b * OFFC + mt * 16 + half * 8)) << 14) + hw;
    #pragma unroll
    for (int r = 0; r < 8; ++r) {
      int m = mt * 16 + half * 8 + r;          // D: M = r + (lane/16)*8
      float bv = bias[m < 108 ? m : 107];      // clamp, padded rows never read
      op[(size_t)r << 14] = acc0[r] + acc1[r] + bv;
    }
  }
}

// ---------------------------------------------------------------------------
// Stage 2: deformable gather + GEMM. One workgroup = (b, group, 32 pixels).
// All 4 groups sample the same x (reference tiles x), only offsets differ.
// ---------------------------------------------------------------------------
__global__ __launch_bounds__(256) void k_dcn(const float* __restrict__ xt,
                                             const float* __restrict__ off,
                                             const _Float16* __restrict__ w2p,
                                             float* __restrict__ ypre) {
  __shared__ _Float16 col[32][COLPITCH];
  int wgid = blockIdx.x;                 // B * 4 * 512
  int b = wgid >> 11;
  int g = (wgid >> 9) & 3;
  int base = (wgid & 511) * 32;
  int h = base >> 7, w0 = base & 127;
  int tid = threadIdx.x, lane = tid & 31, wv = tid >> 5;
  const float* offb = off + ((size_t)(b * OFFC) << 14);

  for (int q = wv * 36; q < wv * 36 + 36; ++q) {
    int pix = q / 9, p = q % 9;
    int hw = base + pix;
    float dy = offb[((size_t)(g * 18 + p * 2)     << 14) + hw];
    float dx = offb[((size_t)(g * 18 + p * 2 + 1) << 14) + hw];
    float mr = offb[((size_t)(72 + g * 9 + p)     << 14) + hw];
    float m  = 1.f / (1.f + expf(-mr));
    float py = (float)h + (float)(p / 3 - 1) + dy;
    float px = (float)(w0 + pix) + (float)(p % 3 - 1) + dx;
    float y0f = floorf(py), x0f = floorf(px);
    float ly = py - y0f, lx = px - x0f;
    int y0 = (int)y0f, x0 = (int)x0f;
    float w00 = (1.f - ly) * (1.f - lx), w01 = (1.f - ly) * lx;
    float w10 = ly * (1.f - lx),         w11 = ly * lx;

    const float* xb = xt + (((size_t)b * HW_) << 6) + 2 * lane;
    float ax = 0.f, ay = 0.f;
    #pragma unroll
    for (int tpt = 0; tpt < 4; ++tpt) {
      int yy = y0 + (tpt >> 1);
      int xx = x0 + (tpt & 1);
      float wt = (tpt == 0) ? w00 : (tpt == 1) ? w01 : (tpt == 2) ? w10 : w11;
      if (yy >= 0 && yy < H_ && xx >= 0 && xx < W_) {
        float2 t = *(const float2*)(xb + (((size_t)(yy * W_ + xx)) << 6));
        ax += wt * t.x;
        ay += wt * t.y;
      }
    }
    int c = 2 * lane;
    col[pix][c * 9 + p]       = (_Float16)(m * ax);
    col[pix][(c + 1) * 9 + p] = (_Float16)(m * ay);
  }
  __syncthreads();

  // 4 m-tiles x 2 n-tiles = 8 tiles, one per wave; dual accumulator chains.
  int mt = wv & 3, nt = wv >> 2;
  v8f acc0 = {}, acc1 = {};
  const _Float16* aw = w2p + (size_t)(g * 4 + mt) * KSTEPS * 512;
  for (int ks = 0; ks < KSTEPS; ks += 2) {
    v16h a0 = afrag(aw, ks, lane);
    v16h b0 = bfrag(&col[0][0], nt, ks, lane);
    v16h a1 = afrag(aw, ks + 1, lane);
    v16h b1 = bfrag(&col[0][0], nt, ks + 1, lane);
    acc0 = __builtin_amdgcn_wmma_f32_16x16x32_f16(false, a0, false, b0,
                                                  (short)0, acc0, false, false);
    acc1 = __builtin_amdgcn_wmma_f32_16x16x32_f16(false, a1, false, b1,
                                                  (short)0, acc1, false, false);
  }
  int n = nt * 16 + (lane & 15);
  int half = lane >> 4;
  int hw = base + n;
  float* op = ypre + (((size_t)((b * 4 + g) * 64 + mt * 16 + half * 8)) << 14) + hw;
  #pragma unroll
  for (int r = 0; r < 8; ++r)
    op[(size_t)r << 14] = acc0[r] + acc1[r];
}

// ---------------------------------------------------------------------------
// Stage 3: per-channel mean/var (BN is over B and the upsampled spatial dims,
// which = all (b,g,hw) at fixed oc). One block per channel.
// ---------------------------------------------------------------------------
__global__ __launch_bounds__(256) void k_bnstats(const float* __restrict__ ypre,
                                                 const float* __restrict__ gamma,
                                                 const float* __restrict__ beta,
                                                 float2* __restrict__ bnp) {
  __shared__ float s1[256], s2[256];
  int oc = blockIdx.x, tid = threadIdx.x;
  float sum = 0.f, sq = 0.f;
  for (int i = tid; i < (int)NCNT; i += 256) {
    int bg = i >> 14, hw = i & 16383;
    float v = ypre[((size_t)(bg * 64 + oc) << 14) + hw];
    sum += v; sq += v * v;
  }
  s1[tid] = sum; s2[tid] = sq;
  __syncthreads();
  for (int s = 128; s > 0; s >>= 1) {
    if (tid < s) { s1[tid] += s1[tid + s]; s2[tid] += s2[tid + s]; }
    __syncthreads();
  }
  if (tid == 0) {
    float mean = s1[0] / NCNT;
    float var  = s2[0] / NCNT - mean * mean;
    float sc   = gamma[oc] * rsqrtf(var + 1e-5f);
    bnp[oc] = make_float2(sc, beta[oc] - mean * sc);
  }
}

// ---------------------------------------------------------------------------
// Stage 4: scale/shift + ReLU with the pixel-shuffle scatter folded into the
// (coalesced) output indexing: out[b,oc,2h+g/2,2w+g%2] = y[b,g,oc,h,w]
// ---------------------------------------------------------------------------
__global__ void k_final(const float* __restrict__ ypre,
                        const float2* __restrict__ bnp,
                        float* __restrict__ out) {
  int t = blockIdx.x * 256 + threadIdx.x;
  if (t >= B_ * 64 * 256 * 256) return;
  int w2 = t & 255, h2 = (t >> 8) & 255, oc = (t >> 16) & 63, b = t >> 22;
  int g = (h2 & 1) * 2 + (w2 & 1);
  int h = h2 >> 1, w = w2 >> 1;
  float v = ypre[((size_t)((b * 4 + g) * 64 + oc) << 14) + (h << 7) + w];
  float2 p = bnp[oc];
  v = v * p.x + p.y;
  out[t] = v > 0.f ? v : 0.f;
}

// ---------------------------------------------------------------------------
extern "C" void kernel_launch(void* const* d_in, const int* in_sizes, int n_in,
                              void* d_out, int out_size, void* d_ws, size_t ws_size,
                              hipStream_t stream) {
  const float* x        = (const float*)d_in[0];
  const float* w_offset = (const float*)d_in[1];
  const float* b_offset = (const float*)d_in[2];
  const float* w_dcn    = (const float*)d_in[3];
  const float* gamma    = (const float*)d_in[4];
  const float* beta     = (const float*)d_in[5];
  float* out = (float*)d_out;

  char* ws = (char*)d_ws;
  // workspace layout (256B-aligned regions), ~57 MB total
  float*    xt   = (float*)(ws);                          //  8,388,608 B
  float*    off  = (float*)(ws + 8388608);                // 14,680,064 B (112 ch padded)
  float*    ypre = (float*)(ws + 23068672);               // 33,554,432 B
  _Float16* w1p  = (_Float16*)(ws + 56623104);            //    129,024 B
  _Float16* w2p  = (_Float16*)(ws + 56752128);            //    294,912 B
  float2*   bnp  = (float2*)(ws + 57047040);              //        512 B

  k_transpose<<<(B_ * HW_ * C_ + 255) / 256, 256, 0, stream>>>(x, xt);
  k_pack_w1<<<(M1TILES * KSTEPS * 512 + 255) / 256, 256, 0, stream>>>(w_offset, w1p);
  k_pack_w2<<<(4 * 4 * KSTEPS * 512 + 255) / 256, 256, 0, stream>>>(w_dcn, w2p);
  k_offconv<<<B_ * 512, 256, 0, stream>>>(xt, w1p, b_offset, off);
  k_dcn<<<B_ * 4 * 512, 256, 0, stream>>>(xt, off, w2p, ypre);
  k_bnstats<<<64, 256, 0, stream>>>(ypre, gamma, beta, bnp);
  k_final<<<(B_ * 64 * 256 * 256 + 255) / 256, 256, 0, stream>>>(ypre, bnp, out);
}
